// ST_LSTM_15599321219552
// MI455X (gfx1250) — matrix operand order repaired and checked by reading, not compiled, not run on hardware
//
#include <hip/hip_runtime.h>
#include <hip/hip_bf16.h>

typedef __bf16 bf16_t;
typedef __attribute__((ext_vector_type(16))) __bf16 v16bf;
typedef __attribute__((ext_vector_type(8)))  float  v8f;
typedef unsigned int u32x4 __attribute__((ext_vector_type(4)));
typedef int          i32x4 __attribute__((ext_vector_type(4)));
typedef int          i32x8 __attribute__((ext_vector_type(8)));

#if defined(__has_builtin)
#if __has_builtin(__builtin_amdgcn_tensor_load_to_lds) && __has_builtin(__builtin_amdgcn_s_wait_tensorcnt)
#define HAVE_TDM 1
#endif
#endif
#ifndef HAVE_TDM
#define HAVE_TDM 0
#endif

#define B_    64
#define TIN_  49
#define NB_   24
#define H_    384
#define SOUT_ 10
#define GN    1920      // 5 gates * H
#define GK    1152      // 3 * H  (x | h_t | h_s)
#define NBLK  30        // persistent blocks (30 * 16 waves = 480 tiles)
#define TPBM  512       // main kernel threads (16 waves)
#define TPB   256       // prep kernels

// LDS: A fragments (4 mt * 36 kt * 1KB) + B fragments (4 ntl * 36 kt * 1KB)
#define SA_BYTES 147456u
#define SB_BYTES 147456u

// ---- workspace layout (bytes) -------------------------------------------
#define OFF_CNT   0u
#define OFF_G     256u                              // gbuf: 64*1920*4 = 491520
#define OFF_HMB   (OFF_G + 491520u)                 // hm bf16: 64*384*2 = 49152
#define OFF_CM    (OFF_HMB + 49152u)                // cm f32 : 98304
#define OFF_HS    (OFF_CM + 98304u)                 // hs bf16 x2: 98304
#define OFF_CS    (OFF_HS + 98304u)                 // cs f32 x2 : 196608
#define OFF_HPREV (OFF_CS + 196608u)                // bf16 [2][24][64][384]: 2359296
#define OFF_CPREV (OFF_HPREV + 2359296u)            // f32  [2][24][64][384]: 4718592
#define OFF_HTMP  (OFF_CPREV + 4718592u)            // f32  [24][64][384]   : 2359296
#define OFF_PBF   (OFF_HTMP + 2359296u)             // bf16 [10][24][64][384]: 11796480
#define OFF_W     (OFF_PBF + 11796480u)             // bf16 swizzled weights: 48*4423680
#define W_RB_BYTES 4423680u                         // per (r,bone): 1152*1920*2

__device__ __forceinline__ float fsig(float x)  { return 1.0f / (1.0f + __expf(-x)); }
__device__ __forceinline__ float ftanh(float x) { return 2.0f / (1.0f + __expf(-2.0f * x)) - 1.0f; }

// -------------------------------------------------------------------------
__global__ void k_zero_cnt(char* ws) {
    if (threadIdx.x == 0 && blockIdx.x == 0) *(unsigned*)(ws + OFF_CNT) = 0u;
}

// per (b,n,h): temporal means -> h_prev/c_prev init (+ concat global_t for r=1)
__global__ void k_state_init(const float* __restrict__ hid, const float* __restrict__ cel,
                             const float* __restrict__ gt, char* __restrict__ ws) {
    int idx = blockIdx.x * blockDim.x + threadIdx.x;          // B*NB*H = 589824
    if (idx >= B_ * NB_ * H_) return;
    int b = idx / (NB_ * H_);
    int rem = idx - b * (NB_ * H_);
    int n = rem / H_, h = rem - n * H_;
    float sh = 0.f, sc = 0.f;
    int base = (b * TIN_ * NB_ + n) * H_ + h;
    for (int t = 0; t < TIN_; ++t) { sh += hid[base + t * NB_ * H_]; sc += cel[base + t * NB_ * H_]; }
    int di = (n * B_ + b) * H_ + h;                           // [NB][B][H]
    float m0 = sh * (1.0f / TIN_);
    float m1 = (sh + gt[(b * NB_ + n) * H_ + h]) * (1.0f / (TIN_ + 1));
    float mc = sc * (1.0f / TIN_);
    bf16_t* hp = (bf16_t*)(ws + OFF_HPREV);
    float*  cp = (float*)(ws + OFF_CPREV);
    float*  ht = (float*)(ws + OFF_HTMP);
    hp[di] = (bf16_t)m0;
    hp[NB_ * B_ * H_ + di] = (bf16_t)m1;
    cp[di] = mc;
    cp[NB_ * B_ * H_ + di] = mc;
    ht[di] = m0;
}

// per (b,h): spatial boundary means over bones
__global__ void k_bound_init(char* __restrict__ ws) {
    int idx = blockIdx.x * blockDim.x + threadIdx.x;          // B*H = 24576
    if (idx >= B_ * H_) return;
    int b = idx / H_, h = idx - b * H_;
    const float* ht = (const float*)(ws + OFF_HTMP);
    const float* cp = (const float*)(ws + OFF_CPREV);
    float sh = 0.f, sc = 0.f;
    for (int n = 0; n < NB_; ++n) {
        int di = (n * B_ + b) * H_ + h;
        sh += ht[di]; sc += cp[di];
    }
    ((bf16_t*)(ws + OFF_HMB))[idx] = (bf16_t)(sh * (1.0f / NB_));
    ((float*)(ws + OFF_CM))[idx] = sc * (1.0f / NB_);
}

// p [B,SOUT,NB,H] fp32 -> pbf [SOUT][NB][B][H] bf16
__global__ void k_p_conv(const float* __restrict__ p, char* __restrict__ ws) {
    int idx = blockIdx.x * blockDim.x + threadIdx.x;          // 5898240
    if (idx >= B_ * SOUT_ * NB_ * H_) return;
    int h = idx % H_; int t = idx / H_;
    int n = t % NB_; t /= NB_;
    int f = t % SOUT_; int b = t / SOUT_;
    ((bf16_t*)(ws + OFF_PBF))[((f * NB_ + n) * B_ + b) * H_ + h] = (bf16_t)p[idx];
}

// U/Wt/Ws fp32 -> bf16, pre-swizzled into 1KB WMMA B-fragments (32B/lane).
// Fragment (nt,kt): lane l holds N = nt*16+(l&15), K = kt*32 + (l>=16?16:0) + 2v + half
__global__ void k_w_conv(const float* __restrict__ U, const float* __restrict__ Wt,
                         const float* __restrict__ Ws, char* __restrict__ ws) {
    unsigned idx = blockIdx.x * blockDim.x + threadIdx.x;     // dest dwords: 53084160
    if (idx >= 48u * (W_RB_BYTES / 4u)) return;
    unsigned rb   = idx / (W_RB_BYTES / 4u);
    unsigned rem  = idx - rb * (W_RB_BYTES / 4u);
    unsigned frag = rem >> 8, d = rem & 255u;
    unsigned lane = d >> 3, v = d & 7u;
    unsigned nt = frag / 36u, kt = frag - nt * 36u;
    unsigned r = rb / NB_, nb = rb - r * NB_;
    unsigned n = nt * 16u + (lane & 15u);
    unsigned g = n / H_, kk = n - g * H_;
    unsigned k0 = kt * 32u + ((lane >= 16u) ? 16u : 0u) + 2u * v;   // even; k0,k0+1 same source
    unsigned s  = k0 / H_;
    unsigned h0 = k0 - s * H_;
    const float* src = (s == 0) ? U : (s == 1) ? Wt : Ws;
    unsigned base = (((r * NB_ + nb) * 5u + g) * H_ + h0) * H_ + kk;
    union { __bf16 h[2]; unsigned u; } pk;
    pk.h[0] = (__bf16)src[base];
    pk.h[1] = (__bf16)src[base + H_];      // h0+1 row
    ((unsigned*)(ws + OFF_W))[idx] = pk.u;
}

// -------------------------------------------------------------------------
__device__ __forceinline__ void grid_sync(unsigned* cnt, unsigned target) {
    __threadfence();
    __syncthreads();
    if (threadIdx.x == 0) {
        atomicAdd(cnt, 1u);
        while (atomicAdd(cnt, 0u) < target) __builtin_amdgcn_s_sleep(2);
    }
    __syncthreads();
    __threadfence();
}

#if HAVE_TDM
// issue one TDM descriptor copying 147456 contiguous bytes global -> LDS
__device__ __forceinline__ void tdm_load_b(const char* gsrc, unsigned lds_off) {
    unsigned long long ga = (unsigned long long)(uintptr_t)gsrc;
    u32x4 g0 = { 1u,                                         // count=1 (valid D#)
                 lds_off,                                    // lds_addr [63:32]
                 (unsigned)(ga & 0xFFFFFFFFu),               // global_addr[31:0]
                 (unsigned)((ga >> 32) & 0x01FFFFFFu) | 0x80000000u }; // [56:32] | type=2
    // 1-D tile: 18432 elements x 8B = 147456B; tensor_dim0=stride=18432, dim1=tile_dim1=1
    i32x8 g1 = { 0x00030000,                                 // data_size=3 (8B), no multicast
                 0x48000000,                                 // tensor_dim0[15:0]=0x4800 << 16
                 0x00010000,                                 // tensor_dim0[31:16]=0 | tensor_dim1=1
                 0x48000000,                                 // tile_dim0=0x4800
                 1,                                          // tile_dim1=1, tile_dim2=0
                 18432, 0, 0 };                              // tensor_dim0_stride
    i32x4 z4 = { 0, 0, 0, 0 };
#if __has_include(<hip/amd_detail/amd_gfx1250_TDM.h>)
    i32x8 z8 = { 0, 0, 0, 0, 0, 0, 0, 0 };
    __builtin_amdgcn_tensor_load_to_lds(g0, g1, z4, z4, z8, 0);   // 6-arg toolchain
#else
    __builtin_amdgcn_tensor_load_to_lds(g0, g1, z4, z4, 0);       // 5-arg toolchain
#endif
}
#endif

__global__ __launch_bounds__(TPBM) void k_st_lstm(const float* __restrict__ bias,
                                                  float* __restrict__ out,
                                                  char* __restrict__ ws) {
    extern __shared__ char smem[];                 // [A frags 144KB][B frags 144KB]
    unsigned* cnt = (unsigned*)(ws + OFF_CNT);
    float*  gbuf  = (float*)(ws + OFF_G);
    bf16_t* hmb   = (bf16_t*)(ws + OFF_HMB);
    float*  cm    = (float*)(ws + OFF_CM);
    bf16_t* hsb   = (bf16_t*)(ws + OFF_HS);
    float*  csb   = (float*)(ws + OFF_CS);
    bf16_t* hprev = (bf16_t*)(ws + OFF_HPREV);
    float*  cprev = (float*)(ws + OFF_CPREV);
    const bf16_t* pbf = (const bf16_t*)(ws + OFF_PBF);
    const char* wbf   = (const char*)(ws + OFF_W);

    const int tid  = threadIdx.x;
    const int lane = tid & 31;
    const int wave = tid >> 5;                     // 0..15
    const int mt   = wave & 3;                     // M-tile of this wave
    const int ntl  = wave >> 2;                    // local N-tile 0..3
    const int nt   = blockIdx.x * 4 + ntl;         // 0..119
    const int row0 = mt * 16 + ((lane < 16) ? 0 : 8);
    const int col  = nt * 16 + (lane & 15);

    unsigned target = 0;
    for (int f = 0; f < SOUT_; ++f) {
        for (int bone = 0; bone < NB_; ++bone) {
            for (int r = 0; r < 2; ++r) {
                const int rb = r * NB_ + bone;
                const bf16_t* X  = (r == 0) ? (pbf + ((f * NB_ + bone) * B_) * H_) : hsb;
                const bf16_t* HT = hprev + (rb * B_) * H_;
                const bf16_t* HS = (bone == 0) ? hmb : (hsb + r * B_ * H_);
                const bf16_t* gsrc[3] = { X, HT, HS };
                // this block's contiguous 147456B slice of pre-swizzled weights
                const char* gB = wbf + (size_t)rb * W_RB_BYTES
                                     + (size_t)blockIdx.x * (4u * 36u * 1024u);

#if HAVE_TDM
                // ---- B fragments: single TDM DMA, overlapped with A staging below
                if (wave == 0)
                    tdm_load_b(gB, __builtin_amdgcn_groupstaticsize() + SA_BYTES);
#endif
                // ---- stage A into LDS, swizzling rows into 1KB WMMA fragments
                // chunk c (16B = 8 elems): s,row,off -> frag ((row/16)*36+s*12+off/4)
                for (int c = tid; c < 9216; c += TPBM) {
                    int s = c / 3072, cc = c - s * 3072;
                    int row = cc / 48, off = cc - row * 48;
                    unsigned loff = (unsigned)(((row >> 4) * 36 + s * 12 + (off >> 2)) * 1024
                                  + (((row & 15) + (((off >> 1) & 1) << 4)) * 32)
                                  + ((off & 1) * 16));
                    *(uint4*)(smem + loff) = *(const uint4*)(gsrc[s] + row * H_ + off * 8);
                }
#if HAVE_TDM
                if (wave == 0) __builtin_amdgcn_s_wait_tensorcnt(0);
#else
                // ---- fallback: stage B fragments with the SIMDs (contiguous copy)
                for (int c = tid; c < 9216; c += TPBM) {
                    *(uint4*)(smem + SA_BYTES + c * 16) = *(const uint4*)(gB + c * 16);
                }
#endif
                __syncthreads();

                // ---- GEMM: one 16x16 tile per wave, operands from LDS
                const char* sA = smem + (unsigned)(mt * 36) * 1024u + (unsigned)lane * 32u;
                const char* sB = smem + SA_BYTES + (unsigned)(ntl * 36) * 1024u
                                      + (unsigned)lane * 32u;
                v8f acc = {};
                #pragma unroll 4
                for (int kt = 0; kt < 36; ++kt) {
                    v16bf Af = *(const v16bf*)(sA + kt * 1024);
                    v16bf Bf = *(const v16bf*)(sB + kt * 1024);
                    acc = __builtin_amdgcn_wmma_f32_16x16x32_bf16(
                              false, Af, false, Bf, (short)0, acc, false, false);
                }

                // ---- prefetch next cell's weight slice toward L2 (hide HBM latency)
                {
                    int r2 = r ^ 1;
                    int bone2 = (r == 1) ? ((bone + 1) % NB_) : bone;
                    const char* gBn = wbf + (size_t)(r2 * NB_ + bone2) * W_RB_BYTES
                                          + (size_t)blockIdx.x * (4u * 36u * 1024u);
                    for (int c = tid; c < 1152; c += TPBM)
                        __builtin_prefetch(gBn + c * 128, 0, 1);
                }

                #pragma unroll
                for (int v = 0; v < 8; ++v) gbuf[(row0 + v) * GN + col] = acc[v];

                target += NBLK; grid_sync(cnt, target);

                // ---- fused gates + state update
                {
                    const float* bb  = bias + (size_t)(rb * 5) * H_;
                    float*  ctp = cprev + (rb * B_) * H_;
                    float*  csp = csb + r * B_ * H_;
                    const float* csr = (bone == 0) ? cm : csp;
                    bf16_t* hsw = hsb + r * B_ * H_;
                    bf16_t* hpw = hprev + (rb * B_) * H_;
                    for (int idx = blockIdx.x * TPBM + tid; idx < B_ * H_; idx += NBLK * TPBM) {
                        int b = idx / H_, k = idx - b * H_;
                        const float* gr = gbuf + b * GN + k;
                        float gi = fsig (gr[0 * H_] + bb[0 * H_ + k]);
                        float gfs= fsig (gr[1 * H_] + bb[1 * H_ + k]);
                        float gft= fsig (gr[2 * H_] + bb[2 * H_ + k]);
                        float go = fsig (gr[3 * H_] + bb[3 * H_ + k]);
                        float gc = ftanh(gr[4 * H_] + bb[4 * H_ + k]);
                        float ch = gi * gc + gft * ctp[idx] + gfs * csr[idx];
                        float hv = go * ftanh(ch);
                        ctp[idx] = ch; csp[idx] = ch;
                        bf16_t hb = (bf16_t)hv;
                        hsw[idx] = hb; hpw[idx] = hb;
                        if (r == 1) {
                            int o = ((b * SOUT_ + f) * NB_ + bone) * H_ + k;
                            out[o] = hv;
                            out[B_ * SOUT_ * NB_ * H_ + o] = ch;
                        }
                    }
                }
                target += NBLK; grid_sync(cnt, target);
            }
        }
    }
}

// -------------------------------------------------------------------------
extern "C" void kernel_launch(void* const* d_in, const int* in_sizes, int n_in,
                              void* d_out, int out_size, void* d_ws, size_t ws_size,
                              hipStream_t stream) {
    const float* hid  = (const float*)d_in[0];
    const float* cel  = (const float*)d_in[1];
    const float* gt   = (const float*)d_in[2];
    // d_in[3] (global_s_state) is unused by the reference
    const float* p    = (const float*)d_in[4];
    const float* U    = (const float*)d_in[5];
    const float* Wt   = (const float*)d_in[6];
    const float* Ws   = (const float*)d_in[7];
    const float* bias = (const float*)d_in[8];
    char* ws = (char*)d_ws;
    float* out = (float*)d_out;

    k_zero_cnt  <<<1, 64, 0, stream>>>(ws);
    k_state_init<<<(B_ * NB_ * H_ + TPB - 1) / TPB, TPB, 0, stream>>>(hid, cel, gt, ws);
    k_bound_init<<<(B_ * H_ + TPB - 1) / TPB, TPB, 0, stream>>>(ws);
    k_p_conv    <<<(B_ * SOUT_ * NB_ * H_ + TPB - 1) / TPB, TPB, 0, stream>>>(p, ws);
    k_w_conv    <<<(48u * (W_RB_BYTES / 4u) + TPB - 1) / TPB, TPB, 0, stream>>>(U, Wt, Ws, ws);
    k_st_lstm   <<<NBLK, TPBM, SA_BYTES + SB_BYTES, stream>>>(bias, out, ws);
}
